// TopKSAE_68805376081992
// MI455X (gfx1250) — matrix operand order repaired and checked by reading, not compiled, not run on hardware
//
#include <hip/hip_runtime.h>
#include <hip/hip_bf16.h>

// ---------------- problem constants (fixed by reference) ----------------
#define B_ROWS 4096
#define D_IN   2048
#define S_SAE  16384
#define TOPK   64
#define EPSF   1e-5f

typedef __bf16 bf16_t;
typedef __attribute__((ext_vector_type(16))) __bf16        v16bf;
typedef __attribute__((ext_vector_type(8)))  float         v8f;
typedef __attribute__((ext_vector_type(4))) unsigned int   uint4v;
typedef __attribute__((ext_vector_type(4))) float          float4v;

// ---------------- CDNA5 async global->LDS path (guarded) ----------------
#if __has_builtin(__builtin_amdgcn_global_load_async_to_lds_b128)
#define ASYNC_LDS 1
// exact parameter type per hipcc diagnostic: vector_size(16) int, addrspace-qualified
typedef int v4i_t __attribute__((vector_size(4 * sizeof(int))));
typedef __attribute__((address_space(1))) v4i_t as1_v4i;
typedef __attribute__((address_space(3))) v4i_t as3_v4i;
typedef __attribute__((address_space(1))) void  as1_void;
typedef __attribute__((address_space(3))) void  as3_void;
__device__ __forceinline__ void async_copy_b128(const void* gsrc, void* ldst) {
    __builtin_amdgcn_global_load_async_to_lds_b128(
        (as1_v4i*)(as1_void*)gsrc, (as3_v4i*)(as3_void*)ldst, 0, 0);
}
#if __has_builtin(__builtin_amdgcn_s_wait_asynccnt)
#define WAIT_ASYNC(n) __builtin_amdgcn_s_wait_asynccnt(n)
#else
#define WAIT_ASYNC(n) asm volatile("s_wait_asynccnt %0" :: "n"(n) : "memory")
#endif
#else
#define ASYNC_LDS 0
#endif

// =====================================================================
// Kernel 1: per-row LayerNorm (ddof=1) + pre_bias fold, split to bf16 hi/lo
// one 256-thread block per row
// =====================================================================
__global__ __launch_bounds__(256)
void prep_ln_kernel(const float* __restrict__ x, const float* __restrict__ pre_bias,
                    bf16_t* __restrict__ aHi, bf16_t* __restrict__ aLo,
                    float* __restrict__ meanArr, float* __restrict__ stdArr)
{
    const int row = blockIdx.x;
    const int tid = threadIdx.x;
    const float* xr = x + (size_t)row * D_IN;

    float vals[8];
    float s1 = 0.f, s2 = 0.f;
#pragma unroll
    for (int i = 0; i < 8; ++i) {
        float v = __builtin_nontemporal_load(xr + tid + i * 256);
        vals[i] = v;
        s1 += v;
        s2 += v * v;
    }
    __shared__ float red1[256], red2[256];
    red1[tid] = s1; red2[tid] = s2;
    __syncthreads();
    for (int off = 128; off > 0; off >>= 1) {
        if (tid < off) { red1[tid] += red1[tid + off]; red2[tid] += red2[tid + off]; }
        __syncthreads();
    }
    __shared__ float sMean, sStd;
    if (tid == 0) {
        float mean = red1[0] * (1.0f / D_IN);
        float var  = (red2[0] - (float)D_IN * mean * mean) * (1.0f / (D_IN - 1));
        if (var < 0.f) var = 0.f;
        float sd = sqrtf(var);
        sMean = mean; sStd = sd;
        meanArr[row] = mean;
        stdArr[row]  = sd;
    }
    __syncthreads();
    const float mean = sMean;
    const float inv  = 1.0f / (sStd + EPSF);
#pragma unroll
    for (int i = 0; i < 8; ++i) {
        int col  = tid + i * 256;
        float a  = (vals[i] - mean) * inv - pre_bias[col];
        bf16_t h = (bf16_t)a;
        bf16_t l = (bf16_t)(a - (float)h);
        size_t o = (size_t)row * D_IN + col;
        aHi[o] = h;
        aLo[o] = l;
    }
}

// =====================================================================
// Kernel 2: W_enc fp32 -> bf16 hi/lo split (grid-stride)
// =====================================================================
__global__ __launch_bounds__(256)
void conv_wenc_kernel(const float* __restrict__ w,
                      bf16_t* __restrict__ hi, bf16_t* __restrict__ lo, size_t n)
{
    size_t i = (size_t)blockIdx.x * 256 + threadIdx.x;
    size_t stride = (size_t)gridDim.x * 256;
    for (; i < n; i += stride) {
        float f  = __builtin_nontemporal_load(w + i);
        bf16_t h = (bf16_t)f;
        hi[i] = h;
        lo[i] = (bf16_t)(f - (float)h);
    }
}

// =====================================================================
// Kernel 3: transpose W_dec [D,S] -> W_decT [S,D] (32x32 LDS tiles)
// =====================================================================
__global__ __launch_bounds__(256)
void transpose_wdec_kernel(const float* __restrict__ wdec, float* __restrict__ wdecT)
{
    __shared__ float tile[32][33];
    const int tx = threadIdx.x & 31;
    const int ty = threadIdx.x >> 5;      // 0..7
    const int sBase = blockIdx.x * 32;    // S direction
    const int dBase = blockIdx.y * 32;    // D direction
#pragma unroll
    for (int i = 0; i < 4; ++i) {
        int d = dBase + ty + i * 8;
        tile[ty + i * 8][tx] = __builtin_nontemporal_load(wdec + (size_t)d * S_SAE + sBase + tx);
    }
    __syncthreads();
#pragma unroll
    for (int i = 0; i < 4; ++i) {
        int s = sBase + ty + i * 8;
        wdecT[(size_t)s * D_IN + dBase + tx] = tile[tx][ty + i * 8];
    }
}

// =====================================================================
// Kernel 4: encode GEMM  preAct[B,S] = A[B,D] * W_enc[S,D]^T + sae_bias
// bf16 hi/lo split: acc = hi*hi + hi*lo + lo*hi (f32 WMMA accumulate).
// WG tile 256(M)x128(N), K-step 32; 8 waves as 4(M)x2(N); wave = 64x64 C
// (4x4 16x16 tiles, 48 WMMA per wave per K-step vs 32 fragment ds_loads).
// Double-buffered LDS staging via GLOBAL_LOAD_ASYNC_TO_LDS_B128 (ASYNCcnt).
// =====================================================================
#define LDS_PITCH 80                       // 64B data + 16B pad per 32-bf16 row
#define AT_ROWS   256                      // A tile rows (M)
#define BT_ROWS   128                      // B tile rows (N)
#define REG_A     (AT_ROWS * LDS_PITCH)    // 20480 B
#define REG_B     (BT_ROWS * LDS_PITCH)    // 10240 B
#define BUF_SZ    (2 * REG_A + 2 * REG_B)  // Ah+Al+Bh+Bl: 61440 B
#define SMEM_SZ   (2 * BUF_SZ)             // double buffer: 122880 B

__global__ __launch_bounds__(256)
void encode_gemm_kernel(const bf16_t* __restrict__ aHi, const bf16_t* __restrict__ aLo,
                        const bf16_t* __restrict__ wHi, const bf16_t* __restrict__ wLo,
                        const float* __restrict__ sae_bias, float* __restrict__ preAct)
{
    extern __shared__ unsigned char smem[];   // SMEM_SZ bytes (dynamic LDS)

    const int tid   = threadIdx.x;
    const int lane  = tid & 31;
    const int wave  = tid >> 5;
    const int waveM = wave >> 1;        // 0..3 -> 64-row band
    const int waveN = wave & 1;         // 0..1 -> 64-col band
    const int l15   = lane & 15;
    const int half  = lane >> 4;
    const int m0 = blockIdx.y * AT_ROWS;
    const int n0 = blockIdx.x * BT_ROWS;

    v8f acc[4][4];
#pragma unroll
    for (int mt = 0; mt < 4; ++mt)
#pragma unroll
        for (int nt = 0; nt < 4; ++nt)
            acc[mt][nt] = (v8f){0.f,0.f,0.f,0.f,0.f,0.f,0.f,0.f};

    union Frag { uint4v q[2]; v16bf v; };

    // stage one K-slice (A: 256x32 hi/lo, B: 128x32 hi/lo) into buffer `buf`
    // 3072 16B chunks total -> 12 per thread
    auto issue_tile = [&](int kk, int buf) {
        const unsigned base = (unsigned)(buf * BUF_SZ);
#pragma unroll
        for (int i = 0; i < 12; ++i) {
            int c = tid + i * 256;
            const bf16_t* g;
            unsigned so;
            if (i < 4) {                       // A-hi: chunks [0,1024)
                int r = c >> 2, ch = c & 3;
                g  = aHi + (size_t)(m0 + r) * D_IN + kk + ch * 8;
                so = base + r * LDS_PITCH + ch * 16;
            } else if (i < 8) {                // A-lo: chunks [1024,2048)
                int cc = c - 1024, r = cc >> 2, ch = cc & 3;
                g  = aLo + (size_t)(m0 + r) * D_IN + kk + ch * 8;
                so = base + REG_A + r * LDS_PITCH + ch * 16;
            } else if (i < 10) {               // B-hi: chunks [2048,2560)
                int cc = c - 2048, r = cc >> 2, ch = cc & 3;
                g  = wHi + (size_t)(n0 + r) * D_IN + kk + ch * 8;
                so = base + 2 * REG_A + r * LDS_PITCH + ch * 16;
            } else {                           // B-lo: chunks [2560,3072)
                int cc = c - 2560, r = cc >> 2, ch = cc & 3;
                g  = wLo + (size_t)(n0 + r) * D_IN + kk + ch * 8;
                so = base + 2 * REG_A + REG_B + r * LDS_PITCH + ch * 16;
            }
#if ASYNC_LDS
            async_copy_b128(g, smem + so);
#else
            *(uint4v*)(smem + so) = *(const uint4v*)g;
#endif
        }
    };

    issue_tile(0, 0);

    int buf = 0;
    for (int kk = 0; kk < D_IN; kk += 32, buf ^= 1) {
        if (kk + 32 < D_IN) {
            issue_tile(kk + 32, buf ^ 1);
#if ASYNC_LDS
            WAIT_ASYNC(12);  // in-order completion: current tile resident
#endif
        } else {
#if ASYNC_LDS
            WAIT_ASYNC(0);
#endif
        }
        __syncthreads();

        const unsigned char* sAh = smem + buf * BUF_SZ;
        const unsigned char* sAl = sAh + REG_A;
        const unsigned char* sBh = sAh + 2 * REG_A;
        const unsigned char* sBl = sAh + 2 * REG_A + REG_B;

        // ---- B fragments (32x16 KxN): lanes 0-15 K=0..7,16..23 of col l15,
        //      lanes 16-31 K=8..15,24..31 -> bytes [half*16,+16) and [32+half*16,+16) ----
        Frag bh[4], bl[4];
#pragma unroll
        for (int nt = 0; nt < 4; ++nt) {
            int rowN = waveN * 64 + nt * 16 + l15;
            int base = rowN * LDS_PITCH + half * 16;
            bh[nt].q[0] = *(const uint4v*)(sBh + base);
            bh[nt].q[1] = *(const uint4v*)(sBh + base + 32);
            bl[nt].q[0] = *(const uint4v*)(sBl + base);
            bl[nt].q[1] = *(const uint4v*)(sBl + base + 32);
        }

        // ---- per mt: A fragments (16x32 MxK), then 12 WMMAs sharing them ----
#pragma unroll
        for (int mt = 0; mt < 4; ++mt) {
            int rowM = waveM * 64 + mt * 16 + l15;
            int base = rowM * LDS_PITCH + half * 16;
            Frag ah, al;
            ah.q[0] = *(const uint4v*)(sAh + base);
            ah.q[1] = *(const uint4v*)(sAh + base + 32);
            al.q[0] = *(const uint4v*)(sAl + base);
            al.q[1] = *(const uint4v*)(sAl + base + 32);
            // group by shared A operand: ah x {bh0..3, bl0..3}, then al x {bh0..3}
#pragma unroll
            for (int nt = 0; nt < 4; ++nt)
                acc[mt][nt] = __builtin_amdgcn_wmma_f32_16x16x32_bf16(
                    false, ah.v, false, bh[nt].v, (short)0, acc[mt][nt], false, false);
#pragma unroll
            for (int nt = 0; nt < 4; ++nt)
                acc[mt][nt] = __builtin_amdgcn_wmma_f32_16x16x32_bf16(
                    false, ah.v, false, bl[nt].v, (short)0, acc[mt][nt], false, false);
#pragma unroll
            for (int nt = 0; nt < 4; ++nt)
                acc[mt][nt] = __builtin_amdgcn_wmma_f32_16x16x32_bf16(
                    false, al.v, false, bh[nt].v, (short)0, acc[mt][nt], false, false);
        }
        __syncthreads();   // all waves done reading `buf` before refill
    }

    // ---- epilogue: C layout (lane<16: M=v, N=lane; lane>=16: M=v+8, N=lane-16) ----
#pragma unroll
    for (int nt = 0; nt < 4; ++nt) {
        int col = n0 + waveN * 64 + nt * 16 + l15;
        float bias = sae_bias[col];
#pragma unroll
        for (int mt = 0; mt < 4; ++mt) {
#pragma unroll
            for (int v = 0; v < 8; ++v) {
                int rowg = m0 + waveM * 64 + mt * 16 + v + half * 8;
                preAct[(size_t)rowg * S_SAE + col] = acc[mt][nt][v] + bias;
            }
        }
    }
}

// =====================================================================
// Kernel 5: exact top-64 per row via 4-level radix histogram on
// order-preserving key = sign-flip transform of float bits.
// =====================================================================
__device__ __forceinline__ unsigned f2key(float f) {
    unsigned u = __float_as_uint(f);
    return (u & 0x80000000u) ? ~u : (u | 0x80000000u);
}

__global__ __launch_bounds__(256)
void topk_kernel(const float* __restrict__ preAct, float* __restrict__ latents,
                 float* __restrict__ tkVal, int* __restrict__ tkIdx)
{
    const int row = blockIdx.x;
    const int tid = threadIdx.x;
    const float* pr = preAct + (size_t)row * S_SAE;

    __shared__ unsigned hist[256];
    __shared__ unsigned sThresh, sNeed, sTie, sCount;

    unsigned prefix = 0, mask = 0, need = TOPK;

    for (int level = 0; level < 4; ++level) {
        hist[tid] = 0;
        __syncthreads();
        const int shift = 24 - level * 8;
        for (int i = tid; i < S_SAE; i += 256) {
            unsigned key = f2key(pr[i]);
            if ((key & mask) == prefix)
                atomicAdd(&hist[(key >> shift) & 0xFFu], 1u);
        }
        __syncthreads();
        if (tid == 0) {
            unsigned cum = 0;
            for (int b = 255; b >= 0; --b) {
                unsigned c = hist[b];
                if (cum + c >= need) { sThresh = (unsigned)b; sNeed = need - cum; break; }
                cum += c;
            }
        }
        __syncthreads();
        prefix |= sThresh << shift;
        mask   |= 0xFFu    << shift;
        need    = sNeed;
        __syncthreads();
    }

    const unsigned T = prefix;      // exact 32-bit k-th key
    const unsigned needEq = need;   // # of ties (== T) to accept
    if (tid == 0) { sTie = 0; sCount = 0; }
    __syncthreads();

    for (int i = tid; i < S_SAE; i += 256) {
        float f = pr[i];
        unsigned key = f2key(f);
        bool take = (key > T);
        if (!take && key == T) {
            unsigned p = atomicAdd(&sTie, 1u);
            take = (p < needEq);
        }
        float rel = f > 0.f ? f : 0.f;
        __builtin_nontemporal_store(take ? rel : 0.f, latents + (size_t)row * S_SAE + i);
        if (take) {
            unsigned slot = atomicAdd(&sCount, 1u);
            tkIdx[row * TOPK + slot] = i;
            tkVal[row * TOPK + slot] = rel;
        }
    }
}

// =====================================================================
// Kernel 6: sparse decode + un-normalize
// recons[row,:] = (sum_j val_j * W_decT[idx_j,:] + pre_bias) * std + mean
// =====================================================================
__global__ __launch_bounds__(256)
void decode_kernel(const float* __restrict__ wdecT,
                   const float* __restrict__ tkVal, const int* __restrict__ tkIdx,
                   const float* __restrict__ pre_bias,
                   const float* __restrict__ meanArr, const float* __restrict__ stdArr,
                   float* __restrict__ recons)
{
    const int row = blockIdx.x;
    const int tid = threadIdx.x;
    __shared__ float sv[TOPK];
    __shared__ int   si[TOPK];
    if (tid < TOPK) {
        sv[tid] = tkVal[row * TOPK + tid];
        si[tid] = tkIdx[row * TOPK + tid];
    }
    __syncthreads();

    const int d0 = tid * 8;
    float acc[8] = {0.f,0.f,0.f,0.f,0.f,0.f,0.f,0.f};
#pragma unroll 4
    for (int j = 0; j < TOPK; ++j) {
        const float* wr = wdecT + (size_t)si[j] * D_IN + d0;
        float4v w0 = *(const float4v*)(wr);
        float4v w1 = *(const float4v*)(wr + 4);
        float v = sv[j];
        acc[0] += v * w0.x; acc[1] += v * w0.y; acc[2] += v * w0.z; acc[3] += v * w0.w;
        acc[4] += v * w1.x; acc[5] += v * w1.y; acc[6] += v * w1.z; acc[7] += v * w1.w;
    }
    const float mean = meanArr[row];
    const float sd   = stdArr[row];
#pragma unroll
    for (int e = 0; e < 8; ++e) {
        int d = d0 + e;
        __builtin_nontemporal_store((acc[e] + pre_bias[d]) * sd + mean,
                                    recons + (size_t)row * D_IN + d);
    }
}

// =====================================================================
// launch
// =====================================================================
extern "C" void kernel_launch(void* const* d_in, const int* in_sizes, int n_in,
                              void* d_out, int out_size, void* d_ws, size_t ws_size,
                              hipStream_t stream)
{
    (void)in_sizes; (void)n_in; (void)out_size; (void)ws_size;

    const float* x      = (const float*)d_in[0];   // [B, D]
    const float* W_enc  = (const float*)d_in[1];   // [S, D]
    const float* W_dec  = (const float*)d_in[2];   // [D, S]
    const float* pre_b  = (const float*)d_in[3];   // [D]
    const float* sae_b  = (const float*)d_in[4];   // [S]
    // d_in[5] = k (fixed at 64 by the reference)

    float* out     = (float*)d_out;
    float* preAct  = out;                                   // [B, S]
    float* latents = out + (size_t)B_ROWS * S_SAE;          // [B, S]
    float* recons  = out + 2 * (size_t)B_ROWS * S_SAE;      // [B, D]

    unsigned char* ws = (unsigned char*)d_ws;
    size_t off = 0;
    bf16_t* aHi   = (bf16_t*)(ws + off); off += (size_t)B_ROWS * D_IN * 2;
    bf16_t* aLo   = (bf16_t*)(ws + off); off += (size_t)B_ROWS * D_IN * 2;
    bf16_t* wHi   = (bf16_t*)(ws + off); off += (size_t)S_SAE  * D_IN * 2;
    bf16_t* wLo   = (bf16_t*)(ws + off); off += (size_t)S_SAE  * D_IN * 2;
    float*  wdecT = (float*) (ws + off); off += (size_t)S_SAE  * D_IN * 4;
    float*  meanA = (float*) (ws + off); off += (size_t)B_ROWS * 4;
    float*  stdA  = (float*) (ws + off); off += (size_t)B_ROWS * 4;
    float*  tkVal = (float*) (ws + off); off += (size_t)B_ROWS * TOPK * 4;
    int*    tkIdx = (int*)   (ws + off); off += (size_t)B_ROWS * TOPK * 4;

    prep_ln_kernel<<<B_ROWS, 256, 0, stream>>>(x, pre_b, aHi, aLo, meanA, stdA);
    conv_wenc_kernel<<<4096, 256, 0, stream>>>(W_enc, wHi, wLo, (size_t)S_SAE * D_IN);
    transpose_wdec_kernel<<<dim3(S_SAE / 32, D_IN / 32), 256, 0, stream>>>(W_dec, wdecT);
    encode_gemm_kernel<<<dim3(S_SAE / BT_ROWS, B_ROWS / AT_ROWS), 256, SMEM_SZ, stream>>>(
        aHi, aLo, wHi, wLo, sae_b, preAct);
    topk_kernel<<<B_ROWS, 256, 0, stream>>>(preAct, latents, tkVal, tkIdx);
    decode_kernel<<<B_ROWS, 256, 0, stream>>>(wdecT, tkVal, tkIdx, pre_b, meanA, stdA, recons);
}
